// self_attention_90975997264107
// MI455X (gfx1250) — matrix operand order repaired and checked by reading, not compiled, not run on hardware
//
#include <hip/hip_runtime.h>

// ---------------------------------------------------------------------------
// Self-attention for MI455X (gfx1250): all GEMMs via v_wmma_f32_16x16x32_bf16.
// ---------------------------------------------------------------------------

typedef __bf16 bf16_t;
typedef __attribute__((ext_vector_type(16))) __bf16 v16bf;
typedef __attribute__((ext_vector_type(8)))  __bf16 v8bf;
typedef __attribute__((ext_vector_type(8)))  float  v8f;

#define DEV __device__ __forceinline__

constexpr int BB  = 2;       // batch
constexpr int S   = 2048;    // seq len
constexpr int D   = 768;     // embed dim
constexpr int H   = 12;      // heads
constexpr int HD  = 64;      // head dim
constexpr int BS  = BB * S;  // 4096 tokens

// workspace layout (bf16 elements)
constexpr size_t OFF_XB  = 0;                              // [BS, D]
constexpr size_t OFF_WT  = OFF_XB  + (size_t)BS * D;       // 4 x [D, D]  (W^T)
constexpr size_t OFF_Q   = OFF_WT  + (size_t)4 * D * D;    // [B,H,S,HD]
constexpr size_t OFF_K   = OFF_Q   + (size_t)BS * D;       // [B,H,S,HD]
constexpr size_t OFF_V   = OFF_K   + (size_t)BS * D;       // [B,H,HD,S] (transposed!)
constexpr size_t OFF_CTX = OFF_V   + (size_t)BS * D;       // [BS, D]
// total = 18,087,936 bf16 = ~34.5 MB

// ---- WMMA fragment loaders (contiguous k = 16*(L>>4) + s, same in A and B) -----

// A-matrix 16x32 bf16: lane L holds row (L&15); elems [0..15]=K base..base+15,
// consecutive, base = (L>>4)*16.  p points at row start (+k off)
DEV v16bf load_afrag(const bf16_t* p, int lane) {
  const int base = (lane >> 4) << 4;
  union { v16bf v; v8bf h[2]; } u;
  u.h[0] = *(const v8bf*)(p + base);
  u.h[1] = *(const v8bf*)(p + base + 8);
  return u.v;
}

// B-matrix 32x16 bf16: lane L holds column (L&15), K = (L>>4)*16 .. +15,
// consecutive.  p points at the 16 consecutive K elements for this lane.
DEV v16bf load_bfrag(const bf16_t* p) {
  union { v16bf v; v8bf h[2]; } u;
  u.h[0] = *(const v8bf*)(p);
  u.h[1] = *(const v8bf*)(p + 8);
  return u.v;
}

DEV v8f wmma_bf16(v16bf a, v16bf b, v8f c) {
  return __builtin_amdgcn_wmma_f32_16x16x32_bf16(
      /*neg_a=*/false, a, /*neg_b=*/false, b,
      /*c_mod=*/(short)0, c, /*reuse_a=*/false, /*reuse_b=*/false);
}

// ---------------------------------------------------------------------------
// Kernel 1: x (f32) -> xb (bf16)
// ---------------------------------------------------------------------------
__global__ void cvt_x_kernel(const float* __restrict__ x, bf16_t* __restrict__ wsb) {
  const int i = blockIdx.x * blockDim.x + threadIdx.x;
  if (i < BS * D) wsb[OFF_XB + i] = (bf16_t)x[i];
}

// ---------------------------------------------------------------------------
// Kernel 2: W{q,k,v,o} (f32 [K,N]) -> W^T (bf16 [N,K]) so B-operand loads are
// contiguous.
// ---------------------------------------------------------------------------
__global__ void cvt_wT_kernel(const float* __restrict__ Wq, const float* __restrict__ Wk,
                              const float* __restrict__ Wv, const float* __restrict__ Wo,
                              bf16_t* __restrict__ wsb) {
  const int gid = blockIdx.x * blockDim.x + threadIdx.x;
  constexpr int PER = D * D;
  if (gid >= 4 * PER) return;
  const int w = gid / PER, e = gid - w * PER;
  const int n = e / D, k = e - n * D;
  const float* src = (w == 0) ? Wq : (w == 1) ? Wk : (w == 2) ? Wv : Wo;
  wsb[OFF_WT + (size_t)w * PER + (size_t)n * D + k] = (bf16_t)src[(size_t)k * D + n];
}

// ---------------------------------------------------------------------------
// Kernel 3: fused QKV projection.  Each wave: one 16x64 tile of one of q/k/v.
// q,k -> [B,H,S,HD] row-major; v -> [B,H,HD,S] transposed.
// ---------------------------------------------------------------------------
__global__ __launch_bounds__(256) void qkv_proj_kernel(
    bf16_t* __restrict__ wsb, const float* __restrict__ bq,
    const float* __restrict__ bk, const float* __restrict__ bv) {
  const int wave = threadIdx.x >> 5, lane = threadIdx.x & 31;
  const int gid  = blockIdx.x * 8 + wave;          // 0..9215
  constexpr int TILES = (BS / 16) * (D / 64);      // 3072 tiles per matrix
  const int mat = gid / TILES;                     // 0=q 1=k 2=v
  const int t   = gid - mat * TILES;
  const int mt  = t / (D / 64);                    // 0..255 (16-row block)
  const int nt  = t - mt * (D / 64);               // 0..11  (== head!)

  const bf16_t* xb = wsb + OFF_XB;
  const bf16_t* WT = wsb + OFF_WT + (size_t)mat * D * D;
  const float*  bias = (mat == 0) ? bq : (mat == 1) ? bk : bv;

  const int col   = lane & 15;
  const int half  = lane >> 4;
  const int khalf = half * 16;
  const bf16_t* arow = xb + (size_t)(mt * 16 + col) * D;

  v8f acc[4] = {v8f{}, v8f{}, v8f{}, v8f{}};
  for (int kk = 0; kk < D; kk += 32) {
    const v16bf a = load_afrag(arow + kk, lane);
#pragma unroll
    for (int sbt = 0; sbt < 4; ++sbt) {
      const v16bf b = load_bfrag(WT + (size_t)(nt * 64 + sbt * 16 + col) * D + kk + khalf);
      acc[sbt] = wmma_bf16(a, b, acc[sbt]);
    }
  }

  bf16_t* Qb = wsb + OFF_Q;
  bf16_t* Kb = wsb + OFF_K;
  bf16_t* Vb = wsb + OFF_V;
  const int h = nt;
#pragma unroll
  for (int sbt = 0; sbt < 4; ++sbt) {
    const int d = sbt * 16 + col;
#pragma unroll
    for (int i = 0; i < 8; ++i) {
      const int token = mt * 16 + i + half * 8;      // C layout: row = r + 8*half
      const int b = token >> 11, s = token & (S - 1);
      const bf16_t val = (bf16_t)(acc[sbt][i] + bias[h * 64 + d]);
      if (mat == 2)
        Vb[(((size_t)b * H + h) * HD + d) * S + s] = val;     // transposed
      else if (mat == 0)
        Qb[(((size_t)b * H + h) * S + s) * HD + d] = val;
      else
        Kb[(((size_t)b * H + h) * S + s) * HD + d] = val;
    }
  }
}

// ---------------------------------------------------------------------------
// Kernel 4: flash-style attention.  1 wave per 16-query block; 32 keys/iter.
// ---------------------------------------------------------------------------
__global__ __launch_bounds__(128) void attn_kernel(bf16_t* __restrict__ wsb) {
  const int wave = threadIdx.x >> 5, lane = threadIdx.x & 31;
  const int g  = blockIdx.x * 4 + wave;    // 0..3071
  const int qb = g & 127;                  // query block within head
  const int bh = g >> 7;                   // 0..23 = b*H + h

  const bf16_t* Qh = wsb + OFF_Q + (size_t)bh * S * HD;
  const bf16_t* Kh = wsb + OFF_K + (size_t)bh * S * HD;
  const bf16_t* Vh = wsb + OFF_V + (size_t)bh * HD * S;

  __shared__ __align__(16) bf16_t Pbuf[4][16 * 32];
  bf16_t* pb = Pbuf[wave];

  const int col = lane & 15, half = lane >> 4;

  // Q A-fragments are loop-invariant
  const bf16_t* qrow = Qh + (size_t)(qb * 16 + col) * HD;
  const v16bf qa0 = load_afrag(qrow, lane);        // d = 0..31
  const v16bf qa1 = load_afrag(qrow + 32, lane);   // d = 32..63

  v8f m_run, l_run{}, O[4] = {v8f{}, v8f{}, v8f{}, v8f{}};
#pragma unroll
  for (int i = 0; i < 8; ++i) m_run[i] = -1e30f;

  constexpr float SC = 0.125f;  // 1/sqrt(64)

  for (int k0 = 0; k0 < S; k0 += 32) {
    // ---- scores: two 16x16 tiles (keys k0..+15, k0+16..+31), K-dim = 64 ----
    v8f s0{}, s1{};
    {
      const bf16_t* kr0 = Kh + (size_t)(k0 + col) * HD + half * 16;
      s0 = wmma_bf16(qa0, load_bfrag(kr0), s0);
      s0 = wmma_bf16(qa1, load_bfrag(kr0 + 32), s0);
      const bf16_t* kr1 = Kh + (size_t)(k0 + 16 + col) * HD + half * 16;
      s1 = wmma_bf16(qa0, load_bfrag(kr1), s1);
      s1 = wmma_bf16(qa1, load_bfrag(kr1 + 32), s1);
    }
#pragma unroll
    for (int i = 0; i < 8; ++i) { s0[i] *= SC; s1[i] *= SC; }

    // ---- online softmax (row = r + 8*half, columns live across 16 lanes) --
    v8f tmax;
#pragma unroll
    for (int i = 0; i < 8; ++i) tmax[i] = fmaxf(s0[i], s1[i]);
#pragma unroll
    for (int mk = 1; mk < 16; mk <<= 1)
#pragma unroll
      for (int i = 0; i < 8; ++i) tmax[i] = fmaxf(tmax[i], __shfl_xor(tmax[i], mk, 32));

    v8f m_new, alpha, psum;
#pragma unroll
    for (int i = 0; i < 8; ++i) {
      m_new[i] = fmaxf(m_run[i], tmax[i]);
      alpha[i] = __expf(m_run[i] - m_new[i]);
      s0[i]    = __expf(s0[i] - m_new[i]);
      s1[i]    = __expf(s1[i] - m_new[i]);
      psum[i]  = s0[i] + s1[i];
    }
#pragma unroll
    for (int mk = 1; mk < 16; mk <<= 1)
#pragma unroll
      for (int i = 0; i < 8; ++i) psum[i] += __shfl_xor(psum[i], mk, 32);
#pragma unroll
    for (int i = 0; i < 8; ++i) {
      l_run[i] = l_run[i] * alpha[i] + psum[i];
      m_run[i] = m_new[i];
    }
#pragma unroll
    for (int tt = 0; tt < 4; ++tt)
#pragma unroll
      for (int i = 0; i < 8; ++i) O[tt][i] *= alpha[i];

    // ---- re-shape P (C-layout) -> A-operand layout through LDS ------------
#pragma unroll
    for (int i = 0; i < 8; ++i) {
      const int r = i + half * 8;
      pb[r * 32 + col]      = (bf16_t)s0[i];
      pb[r * 32 + 16 + col] = (bf16_t)s1[i];
    }
    asm volatile("s_wait_dscnt 0x0" ::: "memory");  // wave-internal LDS RAW
    const v16bf pa = load_afrag(pb + col * 32, lane);  // P[16x32] as A

    // ---- O += P @ V (K=32), V transposed so B loads are contiguous --------
#pragma unroll
    for (int tt = 0; tt < 4; ++tt) {
      const bf16_t* vp = Vh + (size_t)(tt * 16 + col) * S + k0 + half * 16;
      O[tt] = wmma_bf16(pa, load_bfrag(vp), O[tt]);
    }
  }

  // ---- epilogue: normalize, write ctx bf16 [BS, D] ------------------------
  const int b = bh / H, h = bh - b * H;
  bf16_t* ctx = wsb + OFF_CTX;
#pragma unroll
  for (int tt = 0; tt < 4; ++tt)
#pragma unroll
    for (int i = 0; i < 8; ++i) {
      const int token = b * S + qb * 16 + i + half * 8;
      const int feat  = h * 64 + tt * 16 + col;
      ctx[(size_t)token * D + feat] = (bf16_t)(O[tt][i] / l_run[i]);
    }
}

// ---------------------------------------------------------------------------
// Kernel 5: output projection  out = ctx @ Wo + bo  (f32 out)
// ---------------------------------------------------------------------------
__global__ __launch_bounds__(256) void out_proj_kernel(
    const bf16_t* __restrict__ wsb, const float* __restrict__ bo,
    float* __restrict__ out) {
  const int wave = threadIdx.x >> 5, lane = threadIdx.x & 31;
  const int gid  = blockIdx.x * 8 + wave;          // 0..3071
  const int mt   = gid / (D / 64);
  const int nt   = gid - mt * (D / 64);

  const bf16_t* ctx = wsb + OFF_CTX;
  const bf16_t* WT  = wsb + OFF_WT + (size_t)3 * D * D;  // Wo^T

  const int col = lane & 15, half = lane >> 4, khalf = half * 16;
  const bf16_t* arow = ctx + (size_t)(mt * 16 + col) * D;

  v8f acc[4] = {v8f{}, v8f{}, v8f{}, v8f{}};
  for (int kk = 0; kk < D; kk += 32) {
    const v16bf a = load_afrag(arow + kk, lane);
#pragma unroll
    for (int sbt = 0; sbt < 4; ++sbt) {
      const v16bf b = load_bfrag(WT + (size_t)(nt * 64 + sbt * 16 + col) * D + kk + khalf);
      acc[sbt] = wmma_bf16(a, b, acc[sbt]);
    }
  }
#pragma unroll
  for (int sbt = 0; sbt < 4; ++sbt) {
    const int n = nt * 64 + sbt * 16 + col;
#pragma unroll
    for (int i = 0; i < 8; ++i) {
      const int token = mt * 16 + i + half * 8;
      out[(size_t)token * D + n] = acc[sbt][i] + bo[n];
    }
  }
}

// ---------------------------------------------------------------------------
extern "C" void kernel_launch(void* const* d_in, const int* in_sizes, int n_in,
                              void* d_out, int out_size, void* d_ws, size_t ws_size,
                              hipStream_t stream) {
  (void)in_sizes; (void)n_in; (void)out_size; (void)ws_size;
  const float* x  = (const float*)d_in[0];
  const float* Wq = (const float*)d_in[1];
  const float* bq = (const float*)d_in[2];
  const float* Wk = (const float*)d_in[3];
  const float* bk = (const float*)d_in[4];
  const float* Wv = (const float*)d_in[5];
  const float* bv = (const float*)d_in[6];
  const float* Wo = (const float*)d_in[7];
  const float* bo = (const float*)d_in[8];
  bf16_t* wsb = (bf16_t*)d_ws;   // needs ~36.2 MB
  float* out = (float*)d_out;

  cvt_x_kernel<<<(BS * D + 255) / 256, 256, 0, stream>>>(x, wsb);
  cvt_wT_kernel<<<(4 * D * D + 255) / 256, 256, 0, stream>>>(Wq, Wk, Wv, Wo, wsb);
  qkv_proj_kernel<<<1152, 256, 0, stream>>>(wsb, bq, bk, bv);   // 9216 wave-tiles
  attn_kernel<<<768, 128, 0, stream>>>(wsb);                    // 3072 query blocks
  out_proj_kernel<<<384, 256, 0, stream>>>(wsb, bo, out);       // 3072 wave-tiles
}
